// IKM_3908420239989
// MI455X (gfx1250) — matrix operand order repaired and checked by reading, not compiled, not run on hardware
//
#include <hip/hip_runtime.h>

#define NB    16
#define NCIN  16
#define NCOUT 16
#define NH    384
#define NW    384
#define LCOL  388   // 1 halo col each side (386 used), padded to 388
#define RPB   4     // output rows per conv block
#define LROWS (RPB + 2)
#define WMOD_BYTES 81920          // 16*5*16*2*2*8 bf16

typedef __attribute__((ext_vector_type(16))) __bf16 v16bf;
typedef __attribute__((ext_vector_type(8)))  float  v8f;
typedef __attribute__((ext_vector_type(4)))  unsigned int v4u;
typedef __attribute__((ext_vector_type(8)))  int v8i;
typedef __attribute__((ext_vector_type(4)))  int v4i;

__device__ __forceinline__ unsigned short f2bf(float f) {
    unsigned u = __builtin_bit_cast(unsigned, f);
    return (unsigned short)((u + 0x7FFFu + ((u >> 16) & 1u)) >> 16);  // RNE
}

// ---------------------------------------------------------------------------
// Kernel 1: one block per (b, cin).
// Pooled binary-mask bins -> softmax -> sigmoid -> att[9], then modulated
// weights in WMMA-A-fragment order:
//   wmod[b][chunk(5)][cout(16)][tapPair(2)][cinHalf(2)][cin8(8)] (bf16)
// tap = 2*chunk + tapPair; tap 9 (chunk4, tapPair1) is zero K-padding.
// ---------------------------------------------------------------------------
__global__ void IKM_attmod_kernel(const float* __restrict__ x,
                                  const float* __restrict__ Wt,
                                  unsigned short* __restrict__ wmod) {
    int b   = blockIdx.x >> 4;
    int cin = blockIdx.x & 15;
    const float* xp = x + (size_t)(b * NCIN + cin) * (NH * NW);

    __shared__ unsigned int bins[9];
    __shared__ float att[9];
    int tid = threadIdx.x;
    if (tid < 9) bins[tid] = 0u;
    __syncthreads();

    unsigned int cnt[9];
#pragma unroll
    for (int bi = 0; bi < 9; ++bi) {
        int br = bi / 3, bc = bi % 3;
        unsigned int c = 0;
        for (int i = tid; i < 128 * 128; i += 256) {
            int r = br * 128 + (i >> 7);
            int w = bc * 128 + (i & 127);
            c += (xp[r * NW + w] >= 0.0f) ? 1u : 0u;
        }
        cnt[bi] = c;
    }
#pragma unroll
    for (int bi = 0; bi < 9; ++bi) atomicAdd(&bins[bi], cnt[bi]);
    __syncthreads();

    if (tid == 0) {
        float a[9], mx = -1e30f;
#pragma unroll
        for (int i = 0; i < 9; ++i) { a[i] = (float)bins[i] * (1.0f / 16384.0f); mx = fmaxf(mx, a[i]); }
        float s = 0.0f;
#pragma unroll
        for (int i = 0; i < 9; ++i) { a[i] = __expf(a[i] - mx); s += a[i]; }
        float inv = 9.0f / s;
#pragma unroll
        for (int i = 0; i < 9; ++i) {
            float t = a[i] * inv - 1.0f;
            att[i] = 1.0f + 1.0f / (1.0f + __expf(-t));
        }
    }
    __syncthreads();

    int hb = cin >> 3, ci = cin & 7;
    if (tid < 144) {
        int m = tid / 9, tap = tid % 9;
        int c = tap >> 1, tp = tap & 1;
        float v = Wt[(m * NCIN + cin) * 9 + tap] * att[tap];
        wmod[((((b * 5 + c) * 16 + m) * 2 + tp) * 2 + hb) * 8 + ci] = f2bf(v);
    } else if (tid < 160) {
        int m = tid - 144;
        wmod[((((b * 5 + 4) * 16 + m) * 2 + 1) * 2 + hb) * 8 + ci] = 0;
    }
}

// ---------------------------------------------------------------------------
// Kernel 1b: NCHW f32 -> NHWC bf16 transpose (one block per (b, h)).
// Makes each image row a contiguous 12 KB block matching the conv kernel's
// LDS [row][col][cin16] layout -> staging becomes a pure copy (TDM-able),
// and the conv pass reads bf16 (half the bytes).
// ---------------------------------------------------------------------------
__global__ void IKM_nhwc_kernel(const float* __restrict__ x,
                                unsigned short* __restrict__ xb) {
    int h = blockIdx.x;
    int b = blockIdx.y;
    __shared__ __align__(16) unsigned short t[NW * 16];
    int tid = threadIdx.x;
    for (int i = tid; i < NCIN * NW; i += 256) {
        int cin = i / NW, w = i - cin * NW;
        float v = x[(((size_t)b * NCIN + cin) * NH + h) * NW + w];
        t[w * 16 + cin] = f2bf(v);
    }
    __syncthreads();
    const uint4* src = (const uint4*)t;
    uint4* dst = (uint4*)(xb + ((size_t)b * NH + h) * (NW * 16));
    for (int i = tid; i < (NW * 16 * 2) / 16; i += 256) dst[i] = src[i];
}

// ---------------------------------------------------------------------------
// TDM row copy: 1-D tile of `nelem` bf16 elements, global -> LDS.
// D# per CDNA5 ISA: group0 {count=1, lds_addr,
// global_addr, type=2}; group1 {data_size=1(2B), tensor_dim0=nelem,
// tensor_dim1=1, tile_dim0=nelem, tile_dim1=0(unused), stride0=nelem}.
// This toolchain exposes the 6-arg builtin:
//   (uint32x4 g0, int32x8 g1, int32x4 g2, int32x4 g3, int32x8, i32 cpol)
// ---------------------------------------------------------------------------
#if __has_builtin(__builtin_amdgcn_tensor_load_to_lds)
#define HAVE_TDM 1
__device__ __forceinline__ void tdm_row_load(unsigned lds_byte_addr,
                                             unsigned long long gaddr,
                                             unsigned nelem) {
    v4u g0;
    g0.x = 1u;                                     // count=1, user mode
    g0.y = lds_byte_addr;                          // bits 63:32
    g0.z = (unsigned)gaddr;                        // bits 95:64
    g0.w = (unsigned)(gaddr >> 32) | (2u << 30);   // addr[56:32] | type=2
    v8i g1;
    g1[0] = (int)(1u << 16);                       // data_size=1 -> 2 bytes
    g1[1] = (int)((nelem & 0xFFFFu) << 16);        // tensor_dim0[15:0]
    g1[2] = (int)((nelem >> 16) | (1u << 16));     // tensor_dim0[31:16] | tensor_dim1=1
    g1[3] = (int)(nelem << 16);                    // tile_dim0
    g1[4] = 0;                                     // tile_dim1=0, tile_dim2=0
    g1[5] = (int)nelem;                            // tensor_dim0_stride[31:0]
    g1[6] = 0;
    g1[7] = 0;
    v4i z4 = {0, 0, 0, 0};
    v8i z8 = {0, 0, 0, 0, 0, 0, 0, 0};
    __builtin_amdgcn_tensor_load_to_lds(g0, g1, z4, z4, z8, 0);
}
#else
#define HAVE_TDM 0
#endif

// ---------------------------------------------------------------------------
// Kernel 2 (primary): one block (256 thr = 8 waves) per (b, 4 output rows).
// Wave 0 issues 6 TENSOR_LOAD_TO_LDS row copies (12 KB each) from NHWC bf16
// into LDS [6][LCOL][16cin]; s_wait_tensorcnt 0 + barrier publishes.
// Each wave: 12 tiles of 16px x 16cout; 5x v_wmma_f32_16x16x32_bf16 per tile
// over K=(tap,cin) padded to 160.
// ---------------------------------------------------------------------------
__global__ void IKM_dyconv_tdm(const unsigned short* __restrict__ xb,
                               const unsigned short* __restrict__ wmod,
                               float* __restrict__ y) {
    extern __shared__ unsigned short xs[];   // [LROWS][LCOL][16]
    int h0 = blockIdx.x * RPB;
    int b  = blockIdx.y;
    int tid = threadIdx.x;

#if HAVE_TDM
    if (tid < 32) {  // wave 0: issue TDM copies for in-range rows
        unsigned xs_base = (unsigned)(unsigned long long)(uintptr_t)&xs[0]; // LDS offset = addr[31:0]
        for (int r = 0; r < LROWS; ++r) {
            int gr = h0 + r - 1;
            if (gr >= 0 && gr < NH) {
                unsigned lds_addr = xs_base + (unsigned)((r * LCOL + 1) * 16 * 2);
                unsigned long long ga =
                    (unsigned long long)(uintptr_t)(xb + ((size_t)b * NH + gr) * (NW * 16));
                tdm_row_load(lds_addr, ga, NW * 16);
            }
        }
        __builtin_amdgcn_s_wait_tensorcnt(0);
    }
#else
    for (int r = 0; r < LROWS; ++r) {
        int gr = h0 + r - 1;
        if (gr >= 0 && gr < NH) {
            const uint4* src = (const uint4*)(xb + ((size_t)b * NH + gr) * (NW * 16));
            uint4* dst = (uint4*)&xs[(r * LCOL + 1) * 16];
            for (int i = tid; i < (NW * 16 * 2) / 16; i += 256) dst[i] = src[i];
        }
    }
#endif
    // zero-fill out-of-image rows (blocks at top/bottom only)
    for (int r = 0; r < LROWS; ++r) {
        int gr = h0 + r - 1;
        if (gr < 0 || gr >= NH) {
            for (int i = tid; i < NW * 16; i += 256) xs[(r * LCOL + 1) * 16 + i] = 0;
        }
    }
    // zero halo cols 0 and 385
    if (tid < LROWS * 32) {
        int r = tid >> 5, rem = tid & 31;
        int cin = rem & 15, col = (rem >> 4) ? 385 : 0;
        xs[(r * LCOL + col) * 16 + cin] = 0;
    }

    int lane  = tid & 31;
    int khalf = lane >> 4;
    int n     = lane & 15;
    int wave  = tid >> 5;

    v16bf afrag[5];
#pragma unroll
    for (int c = 0; c < 5; ++c) {
        const v16bf* ap = (const v16bf*)(wmod + (size_t)(((b * 5 + c) * 16 + n) * 2 + khalf) * 16);
        afrag[c] = *ap;
    }
    __syncthreads();

#pragma unroll
    for (int ti = 0; ti < 12; ++ti) {
        int tile = wave + 8 * ti;
        int hr = tile / 24;
        int tw = tile - hr * 24;
        int w0 = 16 * tw;
        v8f acc = {};
#pragma unroll
        for (int c = 0; c < 5; ++c) {
            int t = 2 * c + khalf;
            t = (t > 8) ? 8 : t;           // tap-9 garbage killed by zero A
            int kh = t / 3;
            int kw = t - kh * 3;
            int col = w0 + n + kw;
            const v16bf* bp = (const v16bf*)(&xs[((hr + kh) * LCOL + col) * 16]);
            acc = __builtin_amdgcn_wmma_f32_16x16x32_bf16(
                false, afrag[c], false, *bp, (short)0, acc, false, false);
        }
        int h = h0 + hr;
#pragma unroll
        for (int i = 0; i < 8; ++i) {
            int m = i + 8 * khalf;
            y[(((size_t)b * NCOUT + m) * NH + h) * NW + w0 + n] = acc[i];
        }
    }
}

// ---------------------------------------------------------------------------
// Kernel 2 (fallback, f32 source; used only if ws too small for NHWC copy)
// ---------------------------------------------------------------------------
__global__ void IKM_dyconv_f32(const float* __restrict__ x,
                               const unsigned short* __restrict__ wmod,
                               float* __restrict__ y) {
    extern __shared__ unsigned short xs[];
    int h0 = blockIdx.x * RPB;
    int b  = blockIdx.y;
    int tid = threadIdx.x;
    const size_t xbase = (size_t)b * NCIN * NH * NW;
    for (int i = tid; i < LROWS * NCIN * NW; i += 256) {
        int r = i / (NCIN * NW);
        int rem = i - r * (NCIN * NW);
        int cin = rem / NW;
        int w = rem - cin * NW;
        int gr = h0 + r - 1;
        float v = (gr >= 0 && gr < NH) ? x[xbase + ((size_t)cin * NH + gr) * NW + w] : 0.0f;
        xs[(r * LCOL + (w + 1)) * 16 + cin] = f2bf(v);
    }
    if (tid < LROWS * 32) {
        int r = tid >> 5, rem = tid & 31;
        int cin = rem & 15, col = (rem >> 4) ? 385 : 0;
        xs[(r * LCOL + col) * 16 + cin] = 0;
    }
    int lane  = tid & 31;
    int khalf = lane >> 4;
    int n     = lane & 15;
    int wave  = tid >> 5;
    v16bf afrag[5];
#pragma unroll
    for (int c = 0; c < 5; ++c) {
        const v16bf* ap = (const v16bf*)(wmod + (size_t)(((b * 5 + c) * 16 + n) * 2 + khalf) * 16);
        afrag[c] = *ap;
    }
    __syncthreads();
#pragma unroll
    for (int ti = 0; ti < 12; ++ti) {
        int tile = wave + 8 * ti;
        int hr = tile / 24;
        int tw = tile - hr * 24;
        int w0 = 16 * tw;
        v8f acc = {};
#pragma unroll
        for (int c = 0; c < 5; ++c) {
            int t = 2 * c + khalf;
            t = (t > 8) ? 8 : t;
            int kh = t / 3;
            int kw = t - kh * 3;
            int col = w0 + n + kw;
            const v16bf* bp = (const v16bf*)(&xs[((hr + kh) * LCOL + col) * 16]);
            acc = __builtin_amdgcn_wmma_f32_16x16x32_bf16(
                false, afrag[c], false, *bp, (short)0, acc, false, false);
        }
        int h = h0 + hr;
#pragma unroll
        for (int i = 0; i < 8; ++i) {
            int m = i + 8 * khalf;
            y[(((size_t)b * NCOUT + m) * NH + h) * NW + w0 + n] = acc[i];
        }
    }
}

extern "C" void kernel_launch(void* const* d_in, const int* in_sizes, int n_in,
                              void* d_out, int out_size, void* d_ws, size_t ws_size,
                              hipStream_t stream) {
    const float* x  = (const float*)d_in[0];
    const float* Wt = (const float*)d_in[1];
    float* y = (float*)d_out;
    unsigned short* wmod = (unsigned short*)d_ws;

    IKM_attmod_kernel<<<dim3(NB * NCIN), 256, 0, stream>>>(x, Wt, wmod);

    size_t xb_bytes = (size_t)NB * NH * NW * 16 * sizeof(unsigned short);  // 75.5 MB
    size_t shmem = (size_t)LROWS * LCOL * 16 * sizeof(unsigned short);     // 74496 B

    if (ws_size >= WMOD_BYTES + xb_bytes) {
        unsigned short* xb = (unsigned short*)((char*)d_ws + WMOD_BYTES);
        IKM_nhwc_kernel<<<dim3(NH, NB), 256, 0, stream>>>(x, xb);
        IKM_dyconv_tdm<<<dim3(NH / RPB, NB), 256, shmem, stream>>>(xb, wmod, y);
    } else {
        IKM_dyconv_f32<<<dim3(NH / RPB, NB), 256, shmem, stream>>>(x, wmod, y);
    }
}
